// Head_32719060861051
// MI455X (gfx1250) — matrix operand order repaired and checked by reading, not compiled, not run on hardware
//
#include <hip/hip_runtime.h>
#include <math.h>

// Problem constants (match reference)
#define D_MODEL 1024
#define HEAD    64
#define BATCH   8
#define SEQ     2048
#define BT      (BATCH * SEQ)   // 16384 rows

typedef __attribute__((ext_vector_type(16))) __bf16 v16bf;
typedef __attribute__((ext_vector_type(2)))  __bf16 v2bf;
typedef __attribute__((ext_vector_type(8)))  float  v8f;
typedef __attribute__((ext_vector_type(4)))  unsigned uint4v;   // 16B, naturally aligned
typedef __attribute__((ext_vector_type(4)))  int      int4v;

// ---------------- bf16 conversion helpers ----------------
#if __has_builtin(__builtin_amdgcn_cvt_pk_bf16_f32)
__device__ __forceinline__ unsigned pack2bf(float lo, float hi) {
    v2bf p = __builtin_amdgcn_cvt_pk_bf16_f32(lo, hi);
    unsigned r; __builtin_memcpy(&r, &p, 4); return r;
}
#else
// RNE round both, then one v_perm_b32 assembles {hi[31:16], lo[31:16]}
__device__ __forceinline__ unsigned pack2bf(float lo, float hi) {
    union { float f; unsigned u; } a, b; a.f = lo; b.f = hi;
    unsigned ul = a.u + 0x7FFFu + ((a.u >> 16) & 1u);
    unsigned uh = b.u + 0x7FFFu + ((b.u >> 16) & 1u);
    return __builtin_amdgcn_perm(uh, ul, 0x07060302u);
}
#endif

// ---------------- async global -> LDS copy (CDNA5 ASYNCcnt path) ----------------
#if __has_builtin(__builtin_amdgcn_global_load_async_to_lds_b128) && \
    __has_builtin(__builtin_amdgcn_s_wait_asynccnt)
#define HAVE_ASYNC_LDS 1
#endif

#ifdef HAVE_ASYNC_LDS
typedef __attribute__((address_space(1))) int4v ga_int4v;
typedef __attribute__((address_space(3))) int4v ls_int4v;
#endif

__device__ __forceinline__ void copy16(const unsigned short* gsrc, unsigned short* ldst) {
#ifdef HAVE_ASYNC_LDS
    __builtin_amdgcn_global_load_async_to_lds_b128(
        (ga_int4v*)(void*)gsrc, (ls_int4v*)(void*)ldst, 0, 0);
#else
    *(uint4v*)ldst = *(const uint4v*)gsrc;
#endif
}
__device__ __forceinline__ void copy_wait() {
#ifdef HAVE_ASYNC_LDS
    __builtin_amdgcn_s_wait_asynccnt(0);
#endif
}

// ---------------- WMMA helpers ----------------
__device__ __forceinline__ v8f wmma_bf16(v16bf a, v16bf b, v8f c) {
    return __builtin_amdgcn_wmma_f32_16x16x32_bf16(
        false, a, false, b, (short)0, c, false, false);
}

// A-fragment (16x32 bf16, row-major src, ld % 8 == 0, base 16B-aligned) -> 2 x b128.
__device__ __forceinline__ v16bf lds_a_frag(const unsigned short* src, int ld) {
    const int lane  = threadIdx.x & 31;
    const int row   = lane & 15;
    const int kbase = (lane >> 4) * 16;
    union { v16bf v; uint4v q[2]; } f;
    f.q[0] = *(const uint4v*)(src + row * ld + kbase);
    f.q[1] = *(const uint4v*)(src + row * ld + 8 + kbase);
    return f.v;
}

// B-fragment where B[k][n] = src[n][k] (src row-major, ld % 8 == 0) -> 2 x b128.
__device__ __forceinline__ v16bf lds_bt_frag(const unsigned short* src, int ld) {
    const int lane  = threadIdx.x & 31;
    const int col   = lane & 15;
    const int kbase = (lane >> 4) * 16;
    union { v16bf v; uint4v q[2]; } f;
    const unsigned short* p = src + col * ld + kbase;
    f.q[0] = *(const uint4v*)(p);
    f.q[1] = *(const uint4v*)(p + 8);
    return f.v;
}

// Pre-swizzled B-fragment straight from global memory: 32 contiguous bytes/lane.
__device__ __forceinline__ v16bf gbl_b_frag(const unsigned short* base) {
    const int lane = threadIdx.x & 31;
    union { v16bf v; uint4v q[2]; } f;
    const unsigned short* p = base + lane * 16;
    f.q[0] = *(const uint4v*)(p);
    f.q[1] = *(const uint4v*)(p + 8);
    return f.v;
}

// ---------------------------------------------------------------------------
// Kernel 0: swizzle W (fp32 [1024 x 64]) * scale into bf16 WMMA B-fragment order.
// Flat: Wf[((kchunk*4 + ntile)*32 + lane)*16 + e], e -> K = (lane>>4)*16 + e.
// For Wq, scale = (1/sqrt(64)) * log2(e): folds both the attention scale and the
// base-2 softmax conversion into the projection.
// ---------------------------------------------------------------------------
__global__ __launch_bounds__(128) void swizzle_w_kernel(
    const float* __restrict__ W, unsigned short* __restrict__ Wf, float scale) {
    const int tile  = blockIdx.x * 4 + (threadIdx.x >> 5);   // 0..127
    const int lane  = threadIdx.x & 31;
    const int kch   = tile >> 2;
    const int ntile = tile & 3;
    const int col   = lane & 15;
    const int kbase = (lane >> 4) * 16;
    unsigned short* dst = Wf + ((size_t)tile * 32 + lane) * 16;
    const float* src = W + (size_t)(kch * 32 + kbase) * HEAD + ntile * 16 + col;
#pragma unroll
    for (int e = 0; e < 16; e += 2)
        *(unsigned*)&dst[e] = pack2bf(src[(size_t)e * HEAD] * scale,
                                      src[(size_t)(e + 1) * HEAD] * scale);
}

// ---------------------------------------------------------------------------
// Kernel 1: projection  Out(bf16) = X(f32)[BT x 1024] * W -> [BT x 64]
// transpose_out: 0 -> row-major [bt][64]; 1 -> [B][HEAD][SEQ] (for V).
// ---------------------------------------------------------------------------
#define PKC 64
#define LDA 72

__global__ __launch_bounds__(128) void proj_kernel(
    const float* __restrict__ X, const unsigned short* __restrict__ Wf,
    unsigned short* __restrict__ Out, int transpose_out) {
    __shared__ alignas(16) unsigned short As[16][LDA];

    const int tid  = threadIdx.x;
    const int wave = tid >> 5;
    const int lane = tid & 31;
    const int row0 = blockIdx.x * 16;

    v8f acc = {};

    for (int kc = 0; kc < D_MODEL; kc += PKC) {
        // Stage 16x64 fp32 -> bf16 pairs; fixed trip count (no exec predication)
#pragma unroll
        for (int j = 0; j < 4; ++j) {
            int p = tid + j * 128;                 // 0..511
            int r = p >> 5, cp = p & 31;
            const float* xp = X + (size_t)(row0 + r) * D_MODEL + kc + cp * 2;
            float x0 = __builtin_nontemporal_load(xp);      // streamed, keep L2 for W/qkv
            float x1 = __builtin_nontemporal_load(xp + 1);
            *(unsigned*)&As[r][cp * 2] = pack2bf(x0, x1);
        }
        __syncthreads();
#pragma unroll
        for (int kk = 0; kk < PKC; kk += 32) {
            int kch = (kc + kk) >> 5;
            v16bf a = lds_a_frag(&As[0][kk], LDA);
            v16bf b = gbl_b_frag(Wf + ((size_t)(kch * 4 + wave)) * 32 * 16);
            acc = wmma_bf16(a, b, acc);
        }
        __syncthreads();
    }

    const int n    = lane & 15;
    const int half = lane >> 4;
    if (transpose_out) {
        // Out[b][hd][t]; rows m,m+1 are adjacent t -> packed dword stores
        const int b0 = row0 / SEQ, t0 = row0 % SEQ;
        unsigned short* orow = Out + ((size_t)b0 * HEAD + wave * 16 + n) * SEQ + t0 + 8 * half;
#pragma unroll
        for (int r = 0; r < 8; r += 2)
            *(unsigned*)(orow + r) = pack2bf(acc[r], acc[r + 1]);
    } else {
        unsigned short* ob = Out + (size_t)(row0 + 8 * half) * HEAD + wave * 16 + n;
#pragma unroll
        for (int r = 0; r < 8; r += 2) {
            unsigned pk = pack2bf(acc[r], acc[r + 1]);
            ob[(size_t)r * HEAD]       = (unsigned short)pk;
            ob[(size_t)(r + 1) * HEAD] = (unsigned short)(pk >> 16);
        }
    }
}

// ---------------------------------------------------------------------------
// Kernel 2: causal flash attention (base-2 online softmax; scale pre-folded
// into q). q,k row-major bf16; v transposed [B][HEAD][SEQ]; out fp32.
// grid (T/64, B); 128 threads; wave w owns query rows [w*16, w*16+16).
// Double-buffered K/Vt tiles staged via async global->LDS.
// ---------------------------------------------------------------------------
#define LDS64 72

__device__ __forceinline__ void stage_kv(
    const unsigned short* Kg, const unsigned short* Vg,
    unsigned short (*Ksb)[LDS64], unsigned short (*Vtb)[LDS64], int tid) {
#pragma unroll
    for (int j = 0; j < 4; ++j) {
        int c = tid + j * 128;             // 0..511 : 16B chunk id
        int r = c >> 3, off = (c & 7) * 8; // row, element offset
        copy16(Kg + (size_t)r * HEAD + off, &Ksb[r][off]);
        copy16(Vg + (size_t)r * SEQ  + off, &Vtb[r][off]);
    }
}

__global__ __launch_bounds__(128) void attn_kernel(
    const unsigned short* __restrict__ Qb,
    const unsigned short* __restrict__ Kb,
    const unsigned short* __restrict__ Vt,
    float* __restrict__ Out) {
    __shared__ alignas(16) unsigned short Ks[2][64][LDS64];    // [buf][key][hd]
    __shared__ alignas(16) unsigned short Vts[2][64][LDS64];   // [buf][hd][key]
    __shared__ alignas(16) unsigned short Ps[4][16][LDS64];    // per-wave P

    const int tid  = threadIdx.x;
    const int wave = tid >> 5;
    const int lane = tid & 31;
    const int n    = lane & 15;
    const int half = lane >> 4;
    const int b    = blockIdx.y;
    const int qt   = blockIdx.x;

    const unsigned short* Kbase = Kb + (size_t)b * SEQ * HEAD;
    const unsigned short* Vbase = Vt + (size_t)b * HEAD * SEQ;

    // Q fragments live in registers for the whole kernel
    const unsigned short* Qrow = Qb + (size_t)(b * SEQ + qt * 64 + wave * 16) * HEAD;
    v16bf qa0 = lds_a_frag(Qrow, HEAD);       // hd 0..31 (global loads)
    v16bf qa1 = lds_a_frag(Qrow + 32, HEAD);  // hd 32..63

    v8f zero = {};
    v8f O[4];
    float m_i[8], l_i[8];
#pragma unroll
    for (int i = 0; i < 4; ++i) O[i] = zero;
#pragma unroll
    for (int r = 0; r < 8; ++r) { m_i[r] = -__builtin_inff(); l_i[r] = 0.f; }

    // Prologue: stage tile 0 into buffer 0
    stage_kv(Kbase, Vbase, Ks[0], Vts[0], tid);
    copy_wait();
    __syncthreads();

    for (int kt = 0; kt <= qt; ++kt) {
        const int cur = kt & 1;
        // Prefetch next tile into the other buffer (overlaps with compute below)
        if (kt < qt)
            stage_kv(Kbase + (size_t)(kt + 1) * 64 * HEAD,
                     Vbase + (size_t)(kt + 1) * 64,
                     Ks[cur ^ 1], Vts[cur ^ 1], tid);

        // Scores (already scaled by 1/sqrt(d)*log2(e) via Wq): S = Q * K^T
        v8f s[4];
#pragma unroll
        for (int t = 0; t < 4; ++t) {
            v8f z = zero;
            z = wmma_bf16(qa0, lds_bt_frag(&Ks[cur][t * 16][0],  LDS64), z);
            z = wmma_bf16(qa1, lds_bt_frag(&Ks[cur][t * 16][32], LDS64), z);
            s[t] = z;
        }

        // Causal mask (diagonal tile only; uniform branch)
        if (kt == qt) {
#pragma unroll
            for (int t = 0; t < 4; ++t)
#pragma unroll
                for (int r = 0; r < 8; ++r) {
                    int qg = wave * 16 + r + 8 * half;
                    int kg = t * 16 + n;
                    if (kg > qg) s[t][r] = -__builtin_inff();
                }
        }

        // Base-2 online softmax (row stats via 16-lane butterflies)
        float mn[8], corr[8];
#pragma unroll
        for (int r = 0; r < 8; ++r) {
            float v = fmaxf(fmaxf(s[0][r], s[1][r]), fmaxf(s[2][r], s[3][r]));
#pragma unroll
            for (int off = 1; off < 16; off <<= 1)
                v = fmaxf(v, __shfl_xor(v, off, 32));
            float nm = fmaxf(m_i[r], v);
            mn[r] = nm;
            corr[r] = __builtin_exp2f(m_i[r] - nm);   // bare v_exp_f32
            m_i[r] = nm;
        }
#pragma unroll
        for (int r = 0; r < 8; ++r) {
            float sum = 0.f;
#pragma unroll
            for (int t = 0; t < 4; ++t) {
                float p = __builtin_exp2f(s[t][r] - mn[r]);
                s[t][r] = p;
                sum += p;
            }
#pragma unroll
            for (int off = 1; off < 16; off <<= 1)
                sum += __shfl_xor(sum, off, 32);
            l_i[r] = l_i[r] * corr[r] + sum;
        }

        // Rescale O accumulators
#pragma unroll
        for (int n4 = 0; n4 < 4; ++n4)
#pragma unroll
            for (int r = 0; r < 8; ++r) O[n4][r] *= corr[r];

        // Spill P (bf16) to LDS for C-layout -> A-layout reshuffle (packed cvt)
#pragma unroll
        for (int t = 0; t < 4; ++t)
#pragma unroll
            for (int r = 0; r < 8; r += 2) {
                unsigned pk = pack2bf(s[t][r], s[t][r + 1]);
                Ps[wave][r + 8 * half][t * 16 + n]     = (unsigned short)pk;
                Ps[wave][r + 1 + 8 * half][t * 16 + n] = (unsigned short)(pk >> 16);
            }
        __syncthreads();

        // O += P(16x64) * V_tile(64x64); P A-frags hoisted, V B-frags contiguous
        v16bf pa0 = lds_a_frag(&Ps[wave][0][0],  LDS64);
        v16bf pa1 = lds_a_frag(&Ps[wave][0][32], LDS64);
#pragma unroll
        for (int n4 = 0; n4 < 4; ++n4) {
            O[n4] = wmma_bf16(pa0, lds_bt_frag(&Vts[cur][n4 * 16][0],  LDS64), O[n4]);
            O[n4] = wmma_bf16(pa1, lds_bt_frag(&Vts[cur][n4 * 16][32], LDS64), O[n4]);
        }

        // Ensure prefetch landed and all waves finished with this tile
        copy_wait();
        __syncthreads();
    }

    // Epilogue: out = O * (1/l)  (fast rcp; output not re-read -> nontemporal)
    float linv[8];
#pragma unroll
    for (int r = 0; r < 8; ++r) linv[r] = __builtin_amdgcn_rcpf(l_i[r]);
    float* Orow = Out + (size_t)(b * SEQ + qt * 64 + wave * 16) * HEAD;
#pragma unroll
    for (int n4 = 0; n4 < 4; ++n4)
#pragma unroll
        for (int r = 0; r < 8; ++r) {
            int m = r + 8 * half;
            __builtin_nontemporal_store(O[n4][r] * linv[r],
                                        &Orow[m * HEAD + n4 * 16 + n]);
        }
}

// ---------------------------------------------------------------------------
// Host-side launcher
// ---------------------------------------------------------------------------
extern "C" void kernel_launch(void* const* d_in, const int* in_sizes, int n_in,
                              void* d_out, int out_size, void* d_ws, size_t ws_size,
                              hipStream_t stream) {
    const float* x_k = (const float*)d_in[0];
    const float* x_q = (const float*)d_in[1];
    const float* x_v = (const float*)d_in[2];
    const float* Wk  = (const float*)d_in[3];
    const float* Wq  = (const float*)d_in[4];
    const float* Wv  = (const float*)d_in[5];
    float* out = (float*)d_out;

    // Workspace: bf16 q,k (row-major), v (transposed), + 3 swizzled weights
    const size_t NQKV = (size_t)BT * HEAD;            // 1M elements each
    const size_t NW   = (size_t)D_MODEL * HEAD;       // 64K elements each
    unsigned short* qws = (unsigned short*)d_ws;
    unsigned short* kws = qws + NQKV;
    unsigned short* vtw = kws + NQKV;
    unsigned short* wfq = vtw + NQKV;
    unsigned short* wfk = wfq + NW;
    unsigned short* wfv = wfk + NW;

    // Fold attention scale (1/sqrt(64)) and softmax base-2 conversion into Wq
    const float qscale = 0.125f * 1.44269504088896340736f;  // * log2(e)
    swizzle_w_kernel<<<32, 128, 0, stream>>>(Wq, wfq, qscale);
    swizzle_w_kernel<<<32, 128, 0, stream>>>(Wk, wfk, 1.0f);
    swizzle_w_kernel<<<32, 128, 0, stream>>>(Wv, wfv, 1.0f);

    dim3 pgrid(BT / 16);
    proj_kernel<<<pgrid, 128, 0, stream>>>(x_q, wfq, qws, 0);
    proj_kernel<<<pgrid, 128, 0, stream>>>(x_k, wfk, kws, 0);
    proj_kernel<<<pgrid, 128, 0, stream>>>(x_v, wfv, vtw, 1);

    dim3 agrid(SEQ / 64, BATCH);
    attn_kernel<<<agrid, 128, 0, stream>>>(qws, kws, vtw, out);
}
